// KNN_75153337745593
// MI455X (gfx1250) — matrix operand order repaired and checked by reading, not compiled, not run on hardware
//
#include <hip/hip_runtime.h>
#include <math.h>

typedef __attribute__((ext_vector_type(16))) __bf16 v16bf;
typedef __attribute__((ext_vector_type(8)))  float  v8f;

#define B_  2
#define M_  8192
#define N_  16384
#define C_  256
#define K_  16
#define NG  (N_/16)   // 1024 column groups of 16
#define NI  (N_/32)   // 512 iteration steps (2 groups each)
#define MG  (M_/16)   // 512 row groups
#define KC  (C_/32)   // 8 K-chunks of 32

// ---------------- workspace layout (bytes) ----------------
static constexpr size_t S2_OFF    = 0;                                  // B*N f32
static constexpr size_t Q2_OFF    = S2_OFF + (size_t)B_ * N_ * 4;       // B*M f32
static constexpr size_t SUP_ELEMS = (size_t)B_ * KC * NG * 32 * 16;     // bf16 elems
static constexpr size_t Q_ELEMS   = (size_t)B_ * KC * MG * 32 * 16;
static constexpr size_t SUPHI_OFF = Q2_OFF + (size_t)B_ * M_ * 4;
static constexpr size_t SUPLO_OFF = SUPHI_OFF + SUP_ELEMS * 2;
static constexpr size_t QHI_OFF   = SUPLO_OFF + SUP_ELEMS * 2;
static constexpr size_t QLO_OFF   = QHI_OFF + Q_ELEMS * 2;
// total ~48.2 MB

__device__ __forceinline__ void split_bf16(float x, __bf16& h, __bf16& l) {
    h = (__bf16)x;
    l = (__bf16)(x - (float)h);
}

// ---------------------------------------------------------------------------
// Prep 1: support -> s2 norms + bf16 hi/lo of (-2*s) in WMMA B-matrix layout.
// B 32x16 bf16 layout: lane = ((kk/8)%2)*16 + col, slot = (kk%8) + (kk>=16)*8.
// Per (b,kc,g) the 32 lanes x 16 bf16 block is 1024 contiguous bytes.
// ---------------------------------------------------------------------------
__global__ void knn_prep_support(const float* __restrict__ sup,
                                 float* __restrict__ s2,
                                 __bf16* __restrict__ supHi,
                                 __bf16* __restrict__ supLo) {
    int tid = blockIdx.x * blockDim.x + threadIdx.x;
    if (tid >= B_ * N_) return;
    int b = tid / N_;
    int n = tid - b * N_;
    const float* s = sup + ((size_t)b * N_ + n) * C_;

    float acc = 0.f;
    int g = n >> 4, col = n & 15;
    for (int k = 0; k < C_; ++k) {
        float v = s[k];
        acc += v * v;
        float x = -2.0f * v;
        __bf16 h, l;
        split_bf16(x, h, l);
        int kc = k >> 5, kk = k & 31;
        int lane = (((kk >> 3) & 1) << 4) | col;
        int slot = (kk & 7) | ((kk >> 4) << 3);
        size_t idx = (((((size_t)b * KC + kc) * NG + g) * 32 + lane) * 16) + slot;
        supHi[idx] = h;
        supLo[idx] = l;
    }
    s2[(size_t)b * N_ + n] = acc;
}

// ---------------------------------------------------------------------------
// Prep 2: query -> q2 norms + bf16 hi/lo in WMMA A-matrix (16x32) layout.
// ---------------------------------------------------------------------------
__global__ void knn_prep_query(const float* __restrict__ qry,
                               float* __restrict__ q2,
                               __bf16* __restrict__ qHi,
                               __bf16* __restrict__ qLo) {
    int tid = blockIdx.x * blockDim.x + threadIdx.x;
    if (tid >= B_ * M_) return;
    int b = tid / M_;
    int m = tid - b * M_;
    const float* q = qry + ((size_t)b * M_ + m) * C_;

    float acc = 0.f;
    int mg = m >> 4, row = m & 15;
    for (int k = 0; k < C_; ++k) {
        float v = q[k];
        acc += v * v;
        __bf16 h, l;
        split_bf16(v, h, l);
        int kc = k >> 5, kk = k & 31;
        int lane, slot;
        if (kk < 8)       { lane = row;      slot = kk;      }
        else if (kk < 16) { lane = row + 16; slot = kk - 8;  }
        else if (kk < 24) { lane = row;      slot = kk - 8;  }
        else              { lane = row + 16; slot = kk - 16; }
        size_t idx = (((((size_t)b * KC + kc) * MG + mg) * 32 + lane) * 16) + slot;
        qHi[idx] = h;
        qLo[idx] = l;
    }
    q2[(size_t)b * M_ + m] = acc;
}

// ---------------------------------------------------------------------------
// Async staging of one 32-column B tile (2 groups, hi+lo, 32 KB) into LDS.
// LDS tile layout: [group-in-pair:16KB][kc:2KB][hi|lo:1KB][lane*32B].
// 128 threads x 16 chunks of 16 B via global_load_async_to_lds_b128.
// ---------------------------------------------------------------------------
__device__ __forceinline__ void issue_async_tile(const char* hiBase,
                                                 const char* loBase,
                                                 int s, unsigned ldsBufByte,
                                                 int tid) {
#pragma unroll
    for (int i = 0; i < 16; ++i) {
        int c   = tid + (i << 7);        // 0..2047: 16-byte chunk id
        int po  = c >> 10;               // group within pair (uniform per i)
        int c2  = c & 1023;
        int kc  = c2 >> 7;               // K-chunk
        int c3  = c2 & 127;
        int arr = c3 >> 6;               // 0 = hi, 1 = lo
        int o   = (c3 & 63) << 4;        // byte offset inside the 1 KB block
        const char* ga = (arr ? loBase : hiBase)
                         + ((size_t)kc * NG + (2 * s + po)) * 1024 + o;
        unsigned la = ldsBufByte + (unsigned)(po << 14) + (unsigned)(kc << 11)
                      + (unsigned)(arr << 10) + (unsigned)o;
        asm volatile("global_load_async_to_lds_b128 %0, %1, off"
                     :: "v"(la), "v"(ga) : "memory");
    }
}

// ---------------------------------------------------------------------------
// Main fused GEMM + top-k kernel.
// 128 threads = 4 waves; wave w owns query rows mg = (mtile*4+w)*16 .. +15.
// Per step: 32 support columns, two independent accumulator chains.
// Ranking key: s2[n] + q . (-2 s[n])  (q2 added after selection).
// ---------------------------------------------------------------------------
__global__ __launch_bounds__(128) void knn_main(
        const float* __restrict__ s2p,
        const float* __restrict__ q2p,
        const __bf16* __restrict__ supHi,
        const __bf16* __restrict__ supLo,
        const __bf16* __restrict__ qHi,
        const __bf16* __restrict__ qLo,
        float* __restrict__ outV,
        int*   __restrict__ outI) {

    __shared__ __align__(16) char ldsB[2 * 32768];   // double-buffered B tiles
    __shared__ float tile[4][16 * 16];
    __shared__ float mrgD[4][32][16];
    __shared__ int   mrgI[4][32][16];

    const int tid  = threadIdx.x;
    const int wave = tid >> 5;
    const int lane = tid & 31;
    const int col  = lane & 15;       // column within 16-group (C-matrix lane)
    const int half = lane >> 4;       // which 8-row / 8-col half
    const int b     = blockIdx.x >> 7;        // 128 mtiles per batch
    const int mtile = blockIdx.x & 127;
    const int mg    = mtile * 4 + wave;       // 16-row group this wave owns

    const char* hiBase = (const char*)supHi + (size_t)b * KC * NG * 1024;
    const char* loBase = (const char*)supLo + (size_t)b * KC * NG * 1024;
    const unsigned ldsB0 = (unsigned)(uintptr_t)&ldsB[0]; // LDS byte offset

    // Load the whole A operand (16 rows x 256 K) hi+lo into registers.
    v16bf aHi[KC], aLo[KC];
#pragma unroll
    for (int kc = 0; kc < KC; ++kc) {
        size_t off = ((((size_t)b * KC + kc) * MG + mg) * 32 + lane) * 16;
        aHi[kc] = *(const v16bf*)(qHi + off);
        aLo[kc] = *(const v16bf*)(qLo + off);
    }

    // Per-lane sorted top-16 (ascending). Lane pair (l, l+16) covers one row.
    float bd[K_];
    int   bi[K_];
#pragma unroll
    for (int i = 0; i < K_; ++i) { bd[i] = 3.0e38f; bi[i] = 0; }

    const float* s2b = s2p + (size_t)b * N_;
    float* tp = tile[wave];

    // Prime the pipeline: tile for step 0 into buffer 0; s2 pair in flight.
    issue_async_tile(hiBase, loBase, 0, ldsB0, tid);
    float s2n0 = s2b[col];
    float s2n1 = s2b[col + 16];

    for (int s = 0; s < NI; ++s) {
        const unsigned curOff = ((unsigned)(s & 1) << 15);
        if (s + 1 < NI) {
            // Prefetch next 32 KB tile into the other buffer (consumed 1
            // step ago; the end-of-step barrier protects the overwrite).
            issue_async_tile(hiBase, loBase, s + 1,
                             ldsB0 + ((unsigned)((s + 1) & 1) << 15), tid);
            asm volatile("s_wait_asynccnt 0x10" ::: "memory"); // step s done
        } else {
            asm volatile("s_wait_asynccnt 0x0" ::: "memory");
        }
        __syncthreads();   // all waves' async chunks for step s are in LDS

        // C init from prefetched s2; kick off the next pair immediately.
        float c0 = s2n0, c1 = s2n1;
        if (s + 1 < NI) {
            s2n0 = s2b[((s + 1) << 5) + col];
            s2n1 = s2b[((s + 1) << 5) + 16 + col];
        }
        v8f acc0 = {c0, c0, c0, c0, c0, c0, c0, c0};
        v8f acc1 = {c1, c1, c1, c1, c1, c1, c1, c1};

        const char* lb = &ldsB[curOff + (lane << 5)];
#pragma unroll
        for (int kc = 0; kc < KC; ++kc) {
            const char* p = lb + (kc << 11);
            v16bf bh0 = *(const v16bf*)p;
            v16bf bl0 = *(const v16bf*)(p + 1024);
            v16bf bh1 = *(const v16bf*)(p + 16384);
            v16bf bl1 = *(const v16bf*)(p + 16384 + 1024);
            acc0 = __builtin_amdgcn_wmma_f32_16x16x32_bf16(
                       false, aHi[kc], false, bh0, (short)0, acc0, false, false);
            acc1 = __builtin_amdgcn_wmma_f32_16x16x32_bf16(
                       false, aHi[kc], false, bh1, (short)0, acc1, false, false);
            acc0 = __builtin_amdgcn_wmma_f32_16x16x32_bf16(
                       false, aLo[kc], false, bh0, (short)0, acc0, false, false);
            acc1 = __builtin_amdgcn_wmma_f32_16x16x32_bf16(
                       false, aLo[kc], false, bh1, (short)0, acc1, false, false);
            acc0 = __builtin_amdgcn_wmma_f32_16x16x32_bf16(
                       false, aHi[kc], false, bl0, (short)0, acc0, false, false);
            acc1 = __builtin_amdgcn_wmma_f32_16x16x32_bf16(
                       false, aHi[kc], false, bl1, (short)0, acc1, false, false);
        }

        // Top-k filter for each of the two 16-column groups.
#pragma unroll
        for (int grp = 0; grp < 2; ++grp) {
            const v8f& acc = grp ? acc1 : acc0;
            // Scatter 16x16 tile to LDS (C layout: vgpr r -> row r + 8*half).
#pragma unroll
            for (int r = 0; r < 8; ++r)
                tp[(((r + (half << 3)) << 4) | col)] = acc[r];
            asm volatile("s_wait_dscnt 0" ::: "memory"); // same-wave LDS RAW

            const float* rp = tp + (col << 4) + (half << 3);
            const int nbase = (s << 5) + (grp << 4) + (half << 3);
#pragma unroll
            for (int j = 0; j < 8; ++j) {
                float d = rp[j];
                if (d < bd[K_ - 1]) {
                    bd[K_ - 1] = d; bi[K_ - 1] = nbase + j;
#pragma unroll
                    for (int t = K_ - 1; t > 0; --t) {
                        if (bd[t] < bd[t - 1]) {
                            float td = bd[t]; bd[t] = bd[t - 1]; bd[t - 1] = td;
                            int   ti = bi[t]; bi[t] = bi[t - 1]; bi[t - 1] = ti;
                        } else break;
                    }
                }
            }
            // Same-wave in-order DS: next scatter cannot pass these reads.
        }
        __syncthreads();   // everyone done reading buffer `curOff`
    }

    // Merge the two per-row lane lists (lane l and l+16) and emit results.
#pragma unroll
    for (int t = 0; t < K_; ++t) { mrgD[wave][lane][t] = bd[t]; mrgI[wave][lane][t] = bi[t]; }
    asm volatile("s_wait_dscnt 0" ::: "memory");

    if (lane < 16) {
        const int m = mg * 16 + lane;
        const float q2v = q2p[(size_t)b * M_ + m];
        size_t obase = ((size_t)b * M_ + m) * K_;
        int ia = 0, ib = 0;
#pragma unroll
        for (int t = 0; t < K_; ++t) {
            float da = (ia < K_) ? mrgD[wave][lane][ia]      : 3.0e38f;
            float db = (ib < K_) ? mrgD[wave][lane + 16][ib] : 3.0e38f;
            bool ta = (da <= db);
            float d = ta ? da : db;
            int  ix = ta ? mrgI[wave][lane][ia] : mrgI[wave][lane + 16][ib];
            ia += ta ? 1 : 0;
            ib += ta ? 0 : 1;
            outV[obase + t] = sqrtf(fmaxf(d + q2v, 0.0f));
            outI[obase + t] = ix;
        }
    }
}

// ---------------------------------------------------------------------------
extern "C" void kernel_launch(void* const* d_in, const int* in_sizes, int n_in,
                              void* d_out, int out_size, void* d_ws, size_t ws_size,
                              hipStream_t stream) {
    const float* query   = (const float*)d_in[0];
    const float* support = (const float*)d_in[1];

    char* ws = (char*)d_ws;
    float*  s2p   = (float*)(ws + S2_OFF);
    float*  q2p   = (float*)(ws + Q2_OFF);
    __bf16* supHi = (__bf16*)(ws + SUPHI_OFF);
    __bf16* supLo = (__bf16*)(ws + SUPLO_OFF);
    __bf16* qHip  = (__bf16*)(ws + QHI_OFF);
    __bf16* qLop  = (__bf16*)(ws + QLO_OFF);

    float* outV = (float*)d_out;
    int*   outI = (int*)d_out + (size_t)B_ * M_ * K_;

    knn_prep_support<<<(B_ * N_ + 255) / 256, 256, 0, stream>>>(support, s2p, supHi, supLo);
    knn_prep_query  <<<(B_ * M_ + 255) / 256, 256, 0, stream>>>(query, q2p, qHip, qLop);

    // 2 batches * 128 row-tiles of 64 rows; 4 waves per block (16 rows/wave).
    knn_main<<<B_ * (M_ / 64), 128, 0, stream>>>(s2p, q2p, supHi, supLo,
                                                 qHip, qLop, outV, outI);
}
